// TorchRoPEAttentionWithLoRA_47124381171924
// MI455X (gfx1250) — matrix operand order repaired and checked by reading, not compiled, not run on hardware
//
#include <hip/hip_runtime.h>

// ---------------------------------------------------------------------------
// Problem constants (from reference)
// ---------------------------------------------------------------------------
#define BB 2
#define SS 2048
#define HH 1024
#define NQ 16
#define NKV 4
#define HD 64
#define RANK 8
#define LORA_SCALE 2.0f          // 16 / RANK
// Q pre-scale = ATTN_SCALE * log2(e) = 0.125 * 1.4426950408889634
// (softmax computed in exp2 domain -> one v_exp_f32 per score, no extra muls)
#define Q_SCALE 0.18033688011112042f

typedef __attribute__((ext_vector_type(8)))  _Float16 v8h;
typedef __attribute__((ext_vector_type(16))) _Float16 v16h;
typedef __attribute__((ext_vector_type(8)))  float    v8f;

static __device__ inline v8f vzero8() {
    v8f z = {0.f,0.f,0.f,0.f,0.f,0.f,0.f,0.f};
    return z;
}

// D = A(16x32 f16) x B(32x16 f16) + C(16x16 f32)
static __device__ inline v8f wmma_f16(v16h a, v16h b, v8f c) {
    return __builtin_amdgcn_wmma_f32_16x16x32_f16(
        /*neg_a=*/false, a, /*neg_b=*/false, b,
        /*c_mod=*/(short)0, c, /*reuse_a=*/false, /*reuse_b=*/false);
}

// A-fragment (16x32, row-major source, 32 contiguous k per row).
// Layout: lanes 0-15 rows, halfs i -> k = hi*16 + i (matches B fragment)
// => 16 contiguous halfs per lane.
static __device__ inline v16h load_a_frag(const _Float16* row32, int hi) {
    v16h a;
    ((v8h*)&a)[0] = *(const v8h*)(row32 + hi * 16);
    ((v8h*)&a)[1] = *(const v8h*)(row32 + hi * 16 + 8);
    return a;
}

// B-fragment (32x16): lane n fixed, k = hi*16 + i => 16 contiguous halfs.
// ptr must already include the hi*16 offset along k.
static __device__ inline v16h load_b_frag(const _Float16* ptr) {
    v16h b;
    ((v8h*)&b)[0] = *(const v8h*)(ptr);
    ((v8h*)&b)[1] = *(const v8h*)(ptr + 8);
    return b;
}

// ---------------------------------------------------------------------------
// Kernel 1: fold LoRA into dense weight, convert to f16.
// out[r][c] = W[r][c] + LORA_SCALE * sum_j Bm[r][j] * A[j][c]
// ---------------------------------------------------------------------------
__global__ void prep_weight(const float* __restrict__ W,
                            const float* __restrict__ A,
                            const float* __restrict__ Bm,
                            _Float16* __restrict__ out,
                            int rows, int cols) {
    int idx = blockIdx.x * blockDim.x + threadIdx.x;
    if (idx >= rows * cols) return;
    int r = idx / cols, c = idx - r * cols;
    float acc = W[idx];
#pragma unroll
    for (int j = 0; j < RANK; ++j)
        acc += LORA_SCALE * Bm[r * RANK + j] * A[j * cols + c];
    out[idx] = (_Float16)acc;
}

// ---------------------------------------------------------------------------
// Kernel 2: f32 -> f16 convert (for x)
// ---------------------------------------------------------------------------
__global__ void cvt_f32_f16(const float* __restrict__ in,
                            _Float16* __restrict__ out, int n) {
    int i = blockIdx.x * blockDim.x + threadIdx.x;
    if (i < n) out[i] = (_Float16)in[i];
}

// ---------------------------------------------------------------------------
// Kernel 3: GEMM  C[M][N] = Ah[M][K] * Wh[N][K]^T   (f16 in, f32 out)
// Block = 128 threads (4 waves, 2x2), wave tile 64x64, k-step 32.
// ---------------------------------------------------------------------------
__global__ __launch_bounds__(128)
void gemm_xwt(const _Float16* __restrict__ Ah,
              const _Float16* __restrict__ Wh,
              float* __restrict__ C, int N, int K) {
    const int lane = threadIdx.x & 31;
    const int wid  = threadIdx.x >> 5;
    const int ln15 = lane & 15;
    const int hi   = lane >> 4;
    const int m0 = blockIdx.x * 128 + (wid & 1) * 64;
    const int n0 = blockIdx.y * 128 + (wid >> 1) * 64;

    v8f acc[4][4];
#pragma unroll
    for (int i = 0; i < 4; ++i)
#pragma unroll
        for (int j = 0; j < 4; ++j) acc[i][j] = vzero8();

    for (int k0 = 0; k0 < K; k0 += 32) {
        v16h a[4], b[4];
#pragma unroll
        for (int mg = 0; mg < 4; ++mg) {
            const _Float16* p = Ah + (size_t)(m0 + mg * 16 + ln15) * K + k0;
            a[mg] = load_a_frag(p, hi);
        }
#pragma unroll
        for (int ng = 0; ng < 4; ++ng) {
            const _Float16* p = Wh + (size_t)(n0 + ng * 16 + ln15) * K + k0 + hi * 16;
            b[ng] = load_b_frag(p);
        }
#pragma unroll
        for (int mg = 0; mg < 4; ++mg)
#pragma unroll
            for (int ng = 0; ng < 4; ++ng)
                acc[mg][ng] = wmma_f16(a[mg], b[ng], acc[mg][ng]);
    }

    // C layout: lane n = L&15, row = (L>>4)*8 + vgpr
#pragma unroll
    for (int mg = 0; mg < 4; ++mg)
#pragma unroll
        for (int ng = 0; ng < 4; ++ng)
#pragma unroll
            for (int r = 0; r < 8; ++r) {
                int m = m0 + mg * 16 + hi * 8 + r;
                int n = n0 + ng * 16 + ln15;
                C[(size_t)m * N + n] = acc[mg][ng][r];
            }
}

// ---------------------------------------------------------------------------
// Kernel 4: RoPE + rearrange into head-major f16 tensors.
// QKV32: [m=b*S+s][1536] f32 raw projections (q:0..1023, k:1024..1279, v:1280..)
// Qh: [b][16][s][64] (pre-scaled by Q_SCALE), Kh: [b][4][s][64],
// Vt: [b][4][64][s] (transposed so P@V B-fragments are contiguous)
// ---------------------------------------------------------------------------
__global__ void rope_arrange(const float* __restrict__ QKV,
                             _Float16* __restrict__ Qh,
                             _Float16* __restrict__ Kh,
                             _Float16* __restrict__ Vt) {
    const int PER = NQ * 32 + NKV * 32 + NKV * 64;   // 896 work items / token
    int id = blockIdx.x * blockDim.x + threadIdx.x;
    if (id >= BB * SS * PER) return;
    int m = id / PER;
    int w = id - m * PER;
    int b = m / SS, s = m - b * SS;
    const float* row = QKV + (size_t)m * 1536;

    if (w < 640) {                       // rope lanes (q: w<512, k: 512..639)
        int h, p;
        float post = 1.0f;               // fold Q scaling into the f32 rope math
        const float* src;
        _Float16* dst;
        if (w < 512) {
            h = w >> 5; p = w & 31;
            src = row + h * HD;
            dst = Qh + ((size_t)(b * NQ + h) * SS + s) * HD;
            post = Q_SCALE;
        } else {
            int w2 = w - 512;
            h = w2 >> 5; p = w2 & 31;
            src = row + 1024 + h * HD;
            dst = Kh + ((size_t)(b * NKV + h) * SS + s) * HD;
        }
        float x1 = src[2 * p], x2 = src[2 * p + 1];
        // inv_freq = 10000^(-p/32)
        float inv = expf(-(float)p * 0.28782313662425574f); // ln(1e4)/32
        float fr = (float)s * inv;
        float sn, cs;
        sincosf(fr, &sn, &cs);
        dst[p]      = (_Float16)((x1 * cs - x2 * sn) * post);
        dst[p + 32] = (_Float16)((x1 * sn + x2 * cs) * post);
    } else {                             // v: plain convert, transposed store
        int w2 = w - 640;
        int hk = w2 >> 6, d = w2 & 63;
        Vt[((size_t)(b * NKV + hk) * HD + d) * SS + s] =
            (_Float16)row[1280 + hk * HD + d];
    }
}

// ---------------------------------------------------------------------------
// Kernel 5: flash attention (full, non-causal), softmax in exp2 domain.
// Block = 128 threads (4 waves); wave owns 32 query rows; WG owns 128 rows.
// Per key tile (64 keys): S = Q*K^T (16 WMMA), online softmax, O += P*V (16 WMMA).
// ---------------------------------------------------------------------------
__global__ __launch_bounds__(128)
void attn_flash(const _Float16* __restrict__ Qh,
                const _Float16* __restrict__ Kh,
                const _Float16* __restrict__ Vt,
                _Float16* __restrict__ Oh) {
    __shared__ _Float16 ldsP[4 * 32 * 64];          // 16 KB, wave-private regions

    const int lane = threadIdx.x & 31;
    const int wid  = threadIdx.x >> 5;
    const int ln15 = lane & 15;
    const int hi   = lane >> 4;
    const int h  = blockIdx.y;
    const int b  = blockIdx.z;
    const int s0 = blockIdx.x * 128 + wid * 32;
    const int hk = h >> 2;                          // GQA: 4 q-heads / kv-head

    const _Float16* Qb = Qh + (size_t)(b * NQ  + h ) * SS * HD;
    const _Float16* Kb = Kh + (size_t)(b * NKV + hk) * SS * HD;
    const _Float16* Vb = Vt + (size_t)(b * NKV + hk) * HD * SS;
    _Float16* lds = ldsP + wid * 32 * 64;

    // Preload Q fragments (already scaled by Q_SCALE in rope_arrange)
    v16h qf[2][2];
#pragma unroll
    for (int g = 0; g < 2; ++g)
#pragma unroll
        for (int kk = 0; kk < 2; ++kk) {
            const _Float16* p = Qb + (size_t)(s0 + g * 16 + ln15) * HD + kk * 32;
            qf[g][kk] = load_a_frag(p, hi);
        }

    v8f oacc[2][4];
#pragma unroll
    for (int g = 0; g < 2; ++g)
#pragma unroll
        for (int oc = 0; oc < 4; ++oc) oacc[g][oc] = vzero8();

    float rmax[2][8], rsum[2][8];
#pragma unroll
    for (int g = 0; g < 2; ++g)
#pragma unroll
        for (int r = 0; r < 8; ++r) { rmax[g][r] = -1e30f; rsum[g][r] = 0.f; }

    for (int t0 = 0; t0 < SS; t0 += 64) {
        if (t0 + 64 < SS) {                         // prefetch next K/V tiles
            __builtin_prefetch(Kb + (size_t)(t0 + 64 + lane * 2) * HD, 0, 1);
            __builtin_prefetch(Vb + (size_t)(lane * 2) * SS + t0 + 64, 0, 1);
        }

        // ---- S = Q * K^T  (2 rowgroups x 4 colgroups of 16 keys) ----
        v8f sacc[2][4];
#pragma unroll
        for (int g = 0; g < 2; ++g)
#pragma unroll
            for (int j = 0; j < 4; ++j) sacc[g][j] = vzero8();

#pragma unroll
        for (int j = 0; j < 4; ++j) {
            v16h kf[2];
#pragma unroll
            for (int kk = 0; kk < 2; ++kk) {
                const _Float16* p =
                    Kb + (size_t)(t0 + j * 16 + ln15) * HD + kk * 32 + hi * 16;
                kf[kk] = load_b_frag(p);
            }
#pragma unroll
            for (int g = 0; g < 2; ++g)
#pragma unroll
                for (int kk = 0; kk < 2; ++kk)
                    sacc[g][j] = wmma_f16(qf[g][kk], kf[kk], sacc[g][j]);
        }

        // ---- online softmax (exp2 domain): row max / rescale ----
        float scale[2][8];
#pragma unroll
        for (int g = 0; g < 2; ++g)
#pragma unroll
            for (int r = 0; r < 8; ++r) {
                float mx = sacc[g][0][r];
                mx = fmaxf(mx, sacc[g][1][r]);
                mx = fmaxf(mx, sacc[g][2][r]);
                mx = fmaxf(mx, sacc[g][3][r]);
                mx = fmaxf(mx, __shfl_xor(mx, 1, 32));
                mx = fmaxf(mx, __shfl_xor(mx, 2, 32));
                mx = fmaxf(mx, __shfl_xor(mx, 4, 32));
                mx = fmaxf(mx, __shfl_xor(mx, 8, 32));
                float nm = fmaxf(rmax[g][r], mx);
                scale[g][r] = exp2f(rmax[g][r] - nm);   // single v_exp_f32
                rmax[g][r]  = nm;
            }

        // ---- P = exp2(S - max): write f16 to LDS, accumulate row sums ----
#pragma unroll
        for (int g = 0; g < 2; ++g)
#pragma unroll
            for (int r = 0; r < 8; ++r) {
                int row = g * 16 + hi * 8 + r;
                float ps = 0.f;
#pragma unroll
                for (int j = 0; j < 4; ++j) {
                    float p = exp2f(sacc[g][j][r] - rmax[g][r]);
                    ps += p;
                    lds[row * 64 + j * 16 + ln15] = (_Float16)p;
                }
                ps += __shfl_xor(ps, 1, 32);
                ps += __shfl_xor(ps, 2, 32);
                ps += __shfl_xor(ps, 4, 32);
                ps += __shfl_xor(ps, 8, 32);
                rsum[g][r] = rsum[g][r] * scale[g][r] + ps;
            }

        // ---- rescale running O ----
#pragma unroll
        for (int g = 0; g < 2; ++g)
#pragma unroll
            for (int oc = 0; oc < 4; ++oc)
#pragma unroll
                for (int r = 0; r < 8; ++r)
                    oacc[g][oc][r] *= scale[g][r];

        // ---- O += P * V: P A-frags loaded once, each V B-frag reused 2x ----
        v16h pf[2][2];
#pragma unroll
        for (int g = 0; g < 2; ++g)
#pragma unroll
            for (int kk = 0; kk < 2; ++kk)
                pf[g][kk] = load_a_frag(lds + (g * 16 + ln15) * 64 + kk * 32, hi);
#pragma unroll
        for (int oc = 0; oc < 4; ++oc)
#pragma unroll
            for (int kk = 0; kk < 2; ++kk) {
                const _Float16* p =
                    Vb + (size_t)(oc * 16 + ln15) * SS + t0 + kk * 32 + hi * 16;
                v16h vf = load_b_frag(p);
                oacc[0][oc] = wmma_f16(pf[0][kk], vf, oacc[0][oc]);
                oacc[1][oc] = wmma_f16(pf[1][kk], vf, oacc[1][oc]);
            }
    }

    // ---- normalize and store Oh[b*S+s][h*64+d] ----
#pragma unroll
    for (int g = 0; g < 2; ++g)
#pragma unroll
        for (int oc = 0; oc < 4; ++oc)
#pragma unroll
            for (int r = 0; r < 8; ++r) {
                int m = b * SS + s0 + g * 16 + hi * 8 + r;
                int n = h * HD + oc * 16 + ln15;
                Oh[(size_t)m * (NQ * HD) + n] =
                    (_Float16)(oacc[g][oc][r] / rsum[g][r]);
            }
}

// ---------------------------------------------------------------------------
// Host launcher
// ---------------------------------------------------------------------------
extern "C" void kernel_launch(void* const* d_in, const int* in_sizes, int n_in,
                              void* d_out, int out_size, void* d_ws, size_t ws_size,
                              hipStream_t stream) {
    const float* x  = (const float*)d_in[0];
    const float* Wq = (const float*)d_in[1];
    const float* Wk = (const float*)d_in[2];
    const float* Wv = (const float*)d_in[3];
    const float* Wo = (const float*)d_in[4];
    const float* qA = (const float*)d_in[5];
    const float* qB = (const float*)d_in[6];
    const float* kA = (const float*)d_in[7];
    const float* kB = (const float*)d_in[8];
    const float* vA = (const float*)d_in[9];
    const float* vB = (const float*)d_in[10];
    const float* oA = (const float*)d_in[11];
    const float* oB = (const float*)d_in[12];
    float* out = (float*)d_out;

    // workspace layout (bytes), total ~59.8 MB, all 256-aligned
    char* ws = (char*)d_ws;
    _Float16* Xh    = (_Float16*)(ws + 0);           //  8.0 MB: x in f16 [4096][1024]
    _Float16* Wqkvh = (_Float16*)(ws + 8388608);     //  3.0 MB: eff qkv W [1536][1024]
    _Float16* Woh   = (_Float16*)(ws + 11534336);    //  2.0 MB: eff o W [1024][1024]
    float*    QKV32 = (float*)   (ws + 13631488);    // 25.2 MB: raw qkv [4096][1536]
    _Float16* Qh    = (_Float16*)(ws + 38797312);    //  8.0 MB: [2][16][2048][64]
    _Float16* Kh    = (_Float16*)(ws + 47185920);    //  2.0 MB: [2][4][2048][64]
    _Float16* Vt    = (_Float16*)(ws + 49283072);    //  2.0 MB: [2][4][64][2048]
    _Float16* Oh    = (_Float16*)(ws + 51380224);    //  8.0 MB: [4096][1024]

    const int M = BB * SS;                           // 4096 tokens

    // 1) convert x -> f16
    cvt_f32_f16<<<(M * HH) / 256, 256, 0, stream>>>(x, Xh, M * HH);

    // 2) fold LoRA into weights (concatenated q|k|v, then o)
    prep_weight<<<(NQ * HD * HH) / 256, 256, 0, stream>>>(Wq, qA, qB, Wqkvh, NQ * HD, HH);
    prep_weight<<<(NKV * HD * HH) / 256, 256, 0, stream>>>(Wk, kA, kB,
                                                           Wqkvh + (size_t)1024 * HH, NKV * HD, HH);
    prep_weight<<<(NKV * HD * HH) / 256, 256, 0, stream>>>(Wv, vA, vB,
                                                           Wqkvh + (size_t)1280 * HH, NKV * HD, HH);
    prep_weight<<<(HH * NQ * HD) / 256, 256, 0, stream>>>(Wo, oA, oB, Woh, HH, NQ * HD);

    // 3) QKV projection: [4096 x 1536] = Xh * Wqkvh^T
    gemm_xwt<<<dim3(M / 128, 1536 / 128), 128, 0, stream>>>(Xh, Wqkvh, QKV32, 1536, HH);

    // 4) RoPE + head-major rearrange (+ V transpose, Q pre-scaled)
    {
        int total = BB * SS * (NQ * 32 + NKV * 32 + NKV * 64);
        rope_arrange<<<(total + 255) / 256, 256, 0, stream>>>(QKV32, Qh, Kh, Vt);
    }

    // 5) flash attention
    attn_flash<<<dim3(SS / 128, NQ, BB), 128, 0, stream>>>(Qh, Kh, Vt, Oh);

    // 6) output projection: out[4096 x 1024] = Oh * Woh^T  (f32 result)
    gemm_xwt<<<dim3(M / 128, HH / 128), 128, 0, stream>>>(Oh, Woh, out, HH, HH);
}
